// CattedLayers_HadamardMLP_37804302139719
// MI455X (gfx1250) — matrix operand order repaired and checked by reading, not compiled, not run on hardware
//
#include <hip/hip_runtime.h>

// ---------------------------------------------------------------------------
// GCN (3 layers, H=D=128) + Hadamard-MLP link predictor + BCE loss, gfx1250.
// GEMMs run on v_wmma_f32_16x16x32_bf16 (bf16 operands, f32 accum).
// Scatter uses native float global atomics; everything else is bandwidth code.
// ---------------------------------------------------------------------------

#define DEV __device__ __forceinline__

typedef __attribute__((ext_vector_type(16))) __bf16        v16bf;
typedef __attribute__((ext_vector_type(8)))  float         v8f;
typedef __attribute__((ext_vector_type(4)))  unsigned int  v4u;
typedef __attribute__((ext_vector_type(8)))  unsigned int  v8u;

union ABPack {          // 32 bytes: 16 bf16 = one WMMA A/B fragment per lane
    v4u   q[2];
    v8u   o;
    v16bf v;
};

DEV unsigned short f2bf(float f) {       // round-to-nearest-even f32 -> bf16
    unsigned int u = __float_as_uint(f);
    u += 0x7FFFu + ((u >> 16) & 1u);
    return (unsigned short)(u >> 16);
}

DEV void atomicAddF(float* p, float v) { // lowers to global_atomic_add_f32
    __hip_atomic_fetch_add(p, v, __ATOMIC_RELAXED, __HIP_MEMORY_SCOPE_AGENT);
}

// --------------------------- small utility kernels -------------------------

__global__ void k_init(float* __restrict__ deg, float* __restrict__ acc, int N) {
    int i = blockIdx.x * 256 + threadIdx.x;
    if (i < N) deg[i] = 1.0f;            // self-loop contributes 1 to degree
    if (i == 0) acc[0] = 0.0f;
}

__global__ void k_deg(const long long* __restrict__ dst, float* __restrict__ deg, int E) {
    int e = blockIdx.x * 256 + threadIdx.x;
    if (e < E) atomicAddF(&deg[(int)dst[e]], 1.0f);
}

__global__ void k_rsqrt(float* __restrict__ dinv, int N) {
    int i = blockIdx.x * 256 + threadIdx.x;
    if (i < N) dinv[i] = rsqrtf(dinv[i]);    // deg >= 1 always
}

__global__ void k_f2bf(const float* __restrict__ in, unsigned short* __restrict__ out, int n) {
    int i = blockIdx.x * 256 + threadIdx.x;
    if (i < n) out[i] = f2bf(in[i]);
}

// Transpose + convert gcn_w: Wt[layer][n][k] = W[layer][k][n]  (bf16)
__global__ void k_wt(const float* __restrict__ W, unsigned short* __restrict__ Wt, int total) {
    int i = blockIdx.x * 256 + threadIdx.x;
    if (i < total) {
        int layer = i >> 14;
        int k     = (i >> 7) & 127;
        int ncol  = i & 127;
        Wt[(layer << 14) + (ncol << 7) + k] = f2bf(W[i]);
    }
}

// --------------------------- WMMA GEMM -------------------------------------
// C[N,128] = A[N,128](bf16) * W[128,128], W given transposed (Bt[n][k]).
// Block = 256 thr = 8 waves; wave w computes 16x16 tile at cols [16w,16w+16).
// K=128 -> 4 x v_wmma_f32_16x16x32_bf16 per wave.
__global__ void __launch_bounds__(256)
k_gemm(const unsigned short* __restrict__ A,
       const unsigned short* __restrict__ Bt,
       float* __restrict__ C, int Nrows) {
    const int wave = threadIdx.x >> 5;
    const int lane = threadIdx.x & 31;
    const int hi   = lane >> 4;          // lane half selects K sub-range
    const int lm   = lane & 15;
    int row = blockIdx.x * 16 + lm;      // A row for this lane
    if (row >= Nrows) row = Nrows - 1;   // keep loads in range; EXEC stays all-1
    const int col = (wave << 4) + lm;    // B/D column for this lane
    const unsigned short* Arow = A  + (size_t)row * 128;
    const unsigned short* Bcol = Bt + (size_t)col * 128;

    v8f c = {};
#pragma unroll
    for (int kk = 0; kk < 4; ++kk) {
        ABPack a, b;
        const int k0 = kk * 32 + hi * 8;
        // A fragment: K = k0..k0+7 (v0-3) and k0+16..k0+23 (v4-7), contiguous pairs
        a.q[0] = *(const v4u*)(Arow + k0);
        a.q[1] = *(const v4u*)(Arow + k0 + 16);
        // B fragment: K = k0..k0+7 and k0+16..k0+23, same order as A (transposed W)
        b.q[0] = *(const v4u*)(Bcol + k0);
        b.q[1] = *(const v4u*)(Bcol + k0 + 16);
        c = __builtin_amdgcn_wmma_f32_16x16x32_bf16(
                false, a.v, false, b.v, (short)0, c, false, false);
    }
    // D layout: vgpr r -> row (8*hi + r), col = lane&15 within tile.
    // Wave-uniform full-tile test: hot path has zero exec-mask churn.
    const int rbase = blockIdx.x * 16 + hi * 8;
    float* Cp = C + (size_t)rbase * 128 + col;
    if ((blockIdx.x + 1) * 16 <= Nrows) {   // full tile (always true for N%16==0)
#pragma unroll
        for (int r = 0; r < 8; ++r)
            Cp[(size_t)r * 128] = c[r];
    } else {                                 // ragged remainder tile
#pragma unroll
        for (int r = 0; r < 8; ++r)
            if (rbase + r < Nrows) Cp[(size_t)r * 128] = c[r];
    }
}

// --------------------------- scatter stage ---------------------------------

// agg = hw * dinv^2   (the self-loop edge, coef = dinv[n]*dinv[n])
__global__ void k_selfinit(const float* __restrict__ hw, const float* __restrict__ dinv,
                           float* __restrict__ agg, int total) {
    int i = blockIdx.x * 256 + threadIdx.x;
    if (i < total) {
        float di = dinv[i >> 7];
        agg[i] = hw[i] * di * di;
    }
}

// One wave per edge: lane handles 4 consecutive feature dims (float4 gather).
__global__ void __launch_bounds__(256)
k_scatter(const long long* __restrict__ src, const long long* __restrict__ dst,
          const float* __restrict__ hw, const float* __restrict__ dinv,
          float* __restrict__ agg, int E) {
    int t = blockIdx.x * 256 + threadIdx.x;
    int e = t >> 5;
    if (e >= E) return;
    int lane = t & 31;
    int s = (int)src[e], d = (int)dst[e];
    float coef = dinv[s] * dinv[d];
    const float4 v = *(const float4*)(hw + (size_t)s * 128 + lane * 4);
    float* out = agg + (size_t)d * 128 + lane * 4;
    atomicAddF(out + 0, v.x * coef);
    atomicAddF(out + 1, v.y * coef);
    atomicAddF(out + 2, v.z * coef);
    atomicAddF(out + 3, v.w * coef);
}

// h = relu(agg + b); write f32 slab of z AND bf16 input for next layer's GEMM
__global__ void k_biasrelu(const float* __restrict__ agg, const float* __restrict__ bias,
                           float* __restrict__ z, unsigned short* __restrict__ Abf,
                           int layer, int total) {
    int i = blockIdx.x * 256 + threadIdx.x;
    if (i < total) {
        int n = i >> 7, d = i & 127;
        float v = fmaxf(agg[i] + bias[layer * 128 + d], 0.0f);
        z[(size_t)n * 384 + layer * 128 + d] = v;
        Abf[i] = f2bf(v);
    }
}

// --------------------------- link predictor + loss -------------------------
// Wave per pair: dot_{d<384} z[a,d]*z[b,d]*pw[d]; softplus(sign*(dot+b)).
__global__ void __launch_bounds__(256)
k_pairloss(const float* __restrict__ z,
           const long long* __restrict__ pa, const long long* __restrict__ pb,
           const float* __restrict__ pw, const float* __restrict__ pbias,
           float sign, int P, float* __restrict__ acc) {
    __shared__ float ssum[8];
    const int wave = threadIdx.x >> 5;
    const int lane = threadIdx.x & 31;
    float wreg[12];
#pragma unroll
    for (int j = 0; j < 12; ++j) wreg[j] = pw[lane + j * 32];
    const float bias = pbias[0];

    float local = 0.0f;
    for (int p = blockIdx.x * 8 + wave; p < P; p += gridDim.x * 8) {
        const float* za = z + (size_t)(int)pa[p] * 384;
        const float* zb = z + (size_t)(int)pb[p] * 384;
        float dot = 0.0f;
#pragma unroll
        for (int j = 0; j < 12; ++j) {
            int d = lane + j * 32;
            dot += za[d] * zb[d] * wreg[j];
        }
#pragma unroll
        for (int off = 16; off > 0; off >>= 1)
            dot += __shfl_down(dot, off, 32);
        if (lane == 0) {
            float x  = sign * (dot + bias);          // -logsigmoid(t)=softplus(-t)
            float sp = (x > 0.0f) ? (x + log1pf(expf(-x))) : log1pf(expf(x));
            local += sp;
        }
    }
    if (lane == 0) ssum[wave] = local;
    __syncthreads();
    if (threadIdx.x == 0) {
        float s = 0.0f;
#pragma unroll
        for (int w = 0; w < 8; ++w) s += ssum[w];
        atomicAddF(acc, s);
    }
}

__global__ void k_finalize(const float* __restrict__ acc, float* __restrict__ out,
                           float invTotal) {
    out[0] = acc[0] * invTotal;
}

// --------------------------- host launch -----------------------------------

extern "C" void kernel_launch(void* const* d_in, const int* in_sizes, int n_in,
                              void* d_out, int out_size, void* d_ws, size_t ws_size,
                              hipStream_t stream) {
    const float*     x   = (const float*)d_in[0];
    const long long* ei  = (const long long*)d_in[1];
    const long long* pos = (const long long*)d_in[2];
    const long long* neg = (const long long*)d_in[3];
    const float*     gw  = (const float*)d_in[4];
    const float*     gb  = (const float*)d_in[5];
    const float*     pw  = (const float*)d_in[6];
    const float*     pb  = (const float*)d_in[7];

    const int N = in_sizes[0] / 128;
    const int E = in_sizes[1] / 2;
    const int P = in_sizes[2] / 2;
    const int L = in_sizes[4] / (128 * 128);   // 3

    auto align256 = [](size_t v) { return (v + 255) & ~(size_t)255; };
    char* w = (char*)d_ws;
    float*          dinv = (float*)w;          w += align256((size_t)N * 4);
    float*          hw   = (float*)w;          w += align256((size_t)N * 128 * 4);
    float*          agg  = (float*)w;          w += align256((size_t)N * 128 * 4);
    float*          z    = (float*)w;          w += align256((size_t)N * 128 * (size_t)L * 4);
    unsigned short* Abf  = (unsigned short*)w; w += align256((size_t)N * 128 * 2);
    unsigned short* Wt   = (unsigned short*)w; w += align256((size_t)L * 128 * 128 * 2);
    float*          acc  = (float*)w;          w += 256;

    const int NE = N * 128;
    auto cdiv = [](int a, int b) { return (a + b - 1) / b; };

    k_init  <<<cdiv(N, 256),            256, 0, stream>>>(dinv, acc, N);
    k_deg   <<<cdiv(E, 256),            256, 0, stream>>>(ei + E, dinv, E);
    k_rsqrt <<<cdiv(N, 256),            256, 0, stream>>>(dinv, N);
    k_f2bf  <<<cdiv(NE, 256),           256, 0, stream>>>(x, Abf, NE);
    k_wt    <<<cdiv(L * 128 * 128, 256),256, 0, stream>>>(gw, Wt, L * 128 * 128);

    for (int layer = 0; layer < L; ++layer) {
        k_gemm     <<<cdiv(N, 16),   256, 0, stream>>>(Abf, Wt + layer * 128 * 128, hw, N);
        k_selfinit <<<cdiv(NE, 256), 256, 0, stream>>>(hw, dinv, agg, NE);
        k_scatter  <<<cdiv(E * 32, 256), 256, 0, stream>>>(ei, ei + E, hw, dinv, agg, E);
        k_biasrelu <<<cdiv(NE, 256), 256, 0, stream>>>(agg, gb, z, Abf, layer, NE);
    }

    k_pairloss <<<512, 256, 0, stream>>>(z, pos, pos + P, pw, pb, -1.0f, P, acc);
    k_pairloss <<<512, 256, 0, stream>>>(z, neg, neg + P, pw, pb, +1.0f, P, acc);
    k_finalize <<<1, 1, 0, stream>>>(acc, (float*)d_out, 1.0f / (2.0f * (float)P));
}
